// PPOLSTMAgent_22110491640581
// MI455X (gfx1250) — matrix operand order repaired and checked by reading, not compiled, not run on hardware
//
#include <hip/hip_runtime.h>
#include <hip/hip_bf16.h>

// ---------------------------------------------------------------------------
// Types for CDNA5 WMMA (wave32): v_wmma_f32_16x16x32_bf16
// ---------------------------------------------------------------------------
typedef __bf16 bf16_t;
typedef bf16_t v16bf __attribute__((ext_vector_type(16)));
typedef bf16_t v8bf  __attribute__((ext_vector_type(8)));
typedef float  v8f   __attribute__((ext_vector_type(8)));

#define T_LEN 128
#define B_ENV 4096
#define N_ROWS (T_LEN * B_ENV)
#define H_DIM 128
#define FEAT_DIM 96
#define NA 6

union AFrag { v16bf v; v8bf h[2]; };

__device__ __forceinline__ v8f wmma_bf16(v16bf a, v16bf b, v8f c) {
  // 8 args: (neg_a, A, neg_b, B, c_mod, C, reuse_a, reuse_b)
  return __builtin_amdgcn_wmma_f32_16x16x32_bf16(false, a, false, b,
                                                 (short)0, c, false, false);
}

// Single-instruction ReLU: v_med3_num_f32 (median of {x, 0, +FLT_MAX}).
__device__ __forceinline__ float reluf_(float x) {
  return __builtin_amdgcn_fmed3f(x, 0.0f, 3.402823466e+38f);
}

// sigmoid via v_exp_f32 + raw v_rcp_f32 (no IEEE div refinement sequence).
__device__ __forceinline__ float sigmoidf_(float x) {
  return __builtin_amdgcn_rcpf(1.0f + __expf(-x));
}

// tanh(x) = 2*sigmoid(2x) - 1  (~5 VALU ops vs libm tanhf; bf16-grade data).
__device__ __forceinline__ float tanhf_(float x) {
  return 2.0f * __builtin_amdgcn_rcpf(1.0f + __expf(-2.0f * x)) - 1.0f;
}

// ---------------------------------------------------------------------------
// Weight prep: f32 -> bf16, transposed into WMMA-B layout Wt[n][k]
// ---------------------------------------------------------------------------
__global__ __launch_bounds__(256) void prep_kernel(
    const float* __restrict__ W1, const float* __restrict__ W2,
    const float* __restrict__ W3, const float* __restrict__ W4,
    const float* __restrict__ Wih, const float* __restrict__ Whh,
    const float* __restrict__ bih, const float* __restrict__ bhh,
    bf16_t* __restrict__ W1t, bf16_t* __restrict__ W2t,
    bf16_t* __restrict__ W3t, bf16_t* __restrict__ W4t,
    bf16_t* __restrict__ Wihb, bf16_t* __restrict__ Whhb,
    float* __restrict__ biasg)
{
  int i = blockIdx.x * 256 + threadIdx.x;
  if (i < 256 * 32) {                       // W1t[256][32], K padded 25->32
    int n = i >> 5, k = i & 31;
    W1t[i] = (bf16_t)((k < 25) ? W1[k * 256 + n] : 0.0f);
    return;
  }
  i -= 256 * 32;
  if (i < 256 * 256) {                      // W2t[256][256]
    int n = i >> 8, k = i & 255;
    W2t[i] = (bf16_t)W2[k * 256 + n];
    return;
  }
  i -= 256 * 256;
  if (i < 128 * 256) {                      // W3t[128][256]
    int n = i >> 8, k = i & 255;
    W3t[i] = (bf16_t)W3[k * 128 + n];
    return;
  }
  i -= 128 * 256;
  if (i < 32 * 128) {                       // W4t[32][128]
    int n = i >> 7, k = i & 127;
    W4t[i] = (bf16_t)W4[k * 32 + n];
    return;
  }
  i -= 32 * 128;
  if (i < 512 * 96)  { Wihb[i] = (bf16_t)Wih[i]; return; }  // already [4H][K]
  i -= 512 * 96;
  if (i < 512 * 128) { Whhb[i] = (bf16_t)Whh[i]; return; }
  i -= 512 * 128;
  if (i < 512)       { biasg[i] = bih[i] + bhh[i]; return; }
}

// ---------------------------------------------------------------------------
// One WMMA GEMM layer over a 64-row LDS tile.
// nt-outer / mtile-inner: B fragments are loaded ONCE per column tile and
// reused across all 4 M-tiles in registers (4x less L2 weight traffic).
// Wave w handles ntiles w, w+8, w+16, ...
// ---------------------------------------------------------------------------
template<int nNT, int nKT>
__device__ __forceinline__ void gemm_layer(
    const bf16_t* in, int lda,
    const bf16_t* __restrict__ wt, int ldw,
    const float* __restrict__ bias,
    bf16_t* out, int ldo,
    int lane, int wave)
{
  const int klo   = (lane >> 4) * 16;   // A-frag: 16 contiguous K per lane
  const int kbsel = (lane >> 4) * 16;   // B-frag: 16 contiguous K per lane
  for (int nt = wave; nt < nNT; nt += 8) {
    const int col = nt * 16 + (lane & 15);
    v16bf bw[nKT];
#pragma unroll
    for (int kt = 0; kt < nKT; ++kt)
      bw[kt] = *(const v16bf*)&wt[(size_t)col * ldw + kt * 32 + kbsel];
    const float bv = bias[col];
#pragma unroll
    for (int m = 0; m < 4; ++m) {
      const int rowA = m * 16 + (lane & 15);
      v8f acc;
#pragma unroll
      for (int r = 0; r < 8; ++r) acc[r] = bv;
#pragma unroll
      for (int kt = 0; kt < nKT; ++kt) {
        AFrag a;
        a.h[0] = *(const v8bf*)&in[rowA * lda + kt * 32 + klo];
        a.h[1] = *(const v8bf*)&in[rowA * lda + kt * 32 + 8 + klo];
        acc = wmma_bf16(a.v, bw[kt], acc);
      }
      const int rbase = m * 16 + 8 * (lane >> 4);
#pragma unroll
      for (int r = 0; r < 8; ++r)
        out[(rbase + r) * ldo + col] = (bf16_t)reluf_(acc[r]);   // v_med3 ReLU
    }
  }
}

// ---------------------------------------------------------------------------
// Fused visual-MLP encoder + loc/energy encoders -> feats[N][96] bf16
// 64 rows per workgroup, whole MLP in LDS.
// ---------------------------------------------------------------------------
__global__ __launch_bounds__(256) void mlp_kernel(
    const float* __restrict__ img, const float* __restrict__ loc,
    const float* __restrict__ eng,
    const float* __restrict__ b1, const float* __restrict__ b2,
    const float* __restrict__ b3, const float* __restrict__ b4,
    const float* __restrict__ Wl, const float* __restrict__ bl,
    const float* __restrict__ We, const float* __restrict__ be,
    const bf16_t* __restrict__ W1t, const bf16_t* __restrict__ W2t,
    const bf16_t* __restrict__ W3t, const bf16_t* __restrict__ W4t,
    bf16_t* __restrict__ feats)
{
  __shared__ bf16_t actA[64 * 256];
  __shared__ bf16_t actB[64 * 256];
  const int tid  = threadIdx.x;
  const int lane = tid & 31;
  const int wave = tid >> 5;
  const int base = blockIdx.x * 64;

  // Stage layer-1 input: x = image/255, K padded 25 -> 32 with zeros.
  for (int idx = tid; idx < 64 * 32; idx += 256) {
    int r = idx >> 5, k = idx & 31;
    float v = (k < 25) ? img[(size_t)(base + r) * 25 + k] * (1.0f / 255.0f)
                       : 0.0f;
    actA[r * 32 + k] = (bf16_t)v;
  }
  __syncthreads();
  gemm_layer<16, 1>(actA, 32,  W1t, 32,  b1, actB, 256, lane, wave); // 25->256
  __syncthreads();
  gemm_layer<16, 8>(actB, 256, W2t, 256, b2, actA, 256, lane, wave); // 256->256
  __syncthreads();
  gemm_layer<8, 8>(actA, 256, W3t, 256, b3, actB, 128, lane, wave);  // 256->128
  __syncthreads();
  gemm_layer<2, 4>(actB, 128, W4t, 128, b4, actA, 32,  lane, wave);  // 128->32
  __syncthreads();

  // feats = [img_feat(32) | loc_feat(32) | eng_feat(32)]
  for (int idx = tid; idx < 64 * 96; idx += 256) {
    int r = idx / 96, cfull = idx % 96;
    size_t n = (size_t)base + r;
    bf16_t v;
    if (cfull < 32) {
      v = actA[r * 32 + cfull];
    } else if (cfull < 64) {
      int j = cfull - 32;
      float x0 = loc[n * 2 + 0] * (1.0f / 10.0f);
      float x1 = loc[n * 2 + 1] * (1.0f / 10.0f);
      v = (bf16_t)(x0 * Wl[j] + x1 * Wl[32 + j] + bl[j]);
    } else {
      int j = cfull - 64;
      v = (bf16_t)(eng[n] * (1.0f / 200.0f) * We[j] + be[j]);
    }
    feats[n * 96 + cfull] = v;
  }
}

// ---------------------------------------------------------------------------
// One LSTM timestep. 64 rows (of B=4096) per workgroup, 64 workgroups.
// gates[B,512] = [x_t | h] @ [Wih;Whh]^T + bias.
// Each wave owns column-quad q so i/f/g/o accumulators for a (row, j) pair
// land in the same lane. kt-outer loop: the 4 gate B-fragments are loaded
// once per k-tile and reused across all 4 M-tiles (register-resident), and
// each kt issues 16 independent WMMAs (hides WMMA->WMMA hazard NOPs).
// ---------------------------------------------------------------------------
__global__ __launch_bounds__(256) void lstm_step_kernel(
    int t, const bf16_t* __restrict__ feats,
    const bf16_t* __restrict__ Wih, const bf16_t* __restrict__ Whh,
    const float* __restrict__ biasg, const int* __restrict__ done,
    float* __restrict__ h, float* __restrict__ c,
    bf16_t* __restrict__ hs)
{
  __shared__ bf16_t hbf[64 * 128];
  const int tid  = threadIdx.x;
  const int lane = tid & 31;
  const int wave = tid >> 5;
  const int base = blockIdx.x * 64;

  // Phase 1: masked previous h -> bf16 in LDS (reset on done).
  for (int idx = tid; idx < 64 * 128; idx += 256) {
    int r = idx >> 7, k = idx & 127;
    int row = base + r;
    float m = 1.0f - (float)done[t * B_ENV + row];
    hbf[idx] = (bf16_t)(h[(size_t)row * 128 + k] * m);
  }
  __syncthreads();

  const int q     = wave;              // column-quad: ntiles q, q+8, q+16, q+24
  const int klo   = (lane >> 4) * 16;
  const int kbsel = (lane >> 4) * 16;
  const int jn    = q * 16 + (lane & 15);   // j within H=128

  // Bias (j is wave-invariant across mtiles): 4 scalars, init 16 accumulators.
  float bg4[4];
#pragma unroll
  for (int g = 0; g < 4; ++g) bg4[g] = biasg[g * 128 + jn];

  v8f acc[4][4];                       // [mtile][gate i,f,g,o]
#pragma unroll
  for (int m = 0; m < 4; ++m)
#pragma unroll
    for (int g = 0; g < 4; ++g)
#pragma unroll
      for (int r = 0; r < 8; ++r) acc[m][g][r] = bg4[g];

#pragma unroll
  for (int kt = 0; kt < 7; ++kt) {     // K = 96 (x) + 128 (h) = 7 * 32
    const int kk = kt * 32 + kbsel;
    v16bf bw[4];
    if (kt < 3) {
#pragma unroll
      for (int g = 0; g < 4; ++g)
        bw[g] = *(const v16bf*)&Wih[(size_t)(jn + g * 128) * 96 + kk];
    } else {
#pragma unroll
      for (int g = 0; g < 4; ++g)
        bw[g] = *(const v16bf*)&Whh[(size_t)(jn + g * 128) * 128 + (kk - 96)];
    }
#pragma unroll
    for (int m = 0; m < 4; ++m) {
      const int rloc = m * 16 + (lane & 15);
      AFrag a;
      if (kt < 3) {
        const bf16_t* p =
            feats + ((size_t)t * B_ENV + (size_t)(base + rloc)) * 96 + kt * 32;
        a.h[0] = *(const v8bf*)(p + klo);
        a.h[1] = *(const v8bf*)(p + 8 + klo);
      } else {
        const bf16_t* p = &hbf[rloc * 128 + (kt - 3) * 32];
        a.h[0] = *(const v8bf*)(p + klo);
        a.h[1] = *(const v8bf*)(p + 8 + klo);
      }
#pragma unroll
      for (int g = 0; g < 4; ++g)
        acc[m][g] = wmma_bf16(a.v, bw[g], acc[m][g]);
    }
  }

  // Elementwise LSTM update — i/f/g/o are co-resident per lane.
#pragma unroll
  for (int m = 0; m < 4; ++m) {
    const int rbase = m * 16 + 8 * (lane >> 4);
#pragma unroll
    for (int r = 0; r < 8; ++r) {
      const int row = base + rbase + r;
      const size_t off = (size_t)row * 128 + jn;
      float mk = 1.0f - (float)done[t * B_ENV + row];
      float cp = c[off] * mk;
      float ig = sigmoidf_(acc[m][0][r]);
      float fg = sigmoidf_(acc[m][1][r]);
      float gg = tanhf_(acc[m][2][r]);
      float og = sigmoidf_(acc[m][3][r]);
      float cn = fg * cp + ig * gg;
      float hn = og * tanhf_(cn);
      c[off] = cn;
      h[off] = hn;
      hs[(size_t)t * ((size_t)B_ENV * 128) + off] = (bf16_t)hn;
    }
  }
}

// ---------------------------------------------------------------------------
// Actor/critic heads: out[n] = [hidden@Wa + ba | hidden@Wc + bc]  (f32)
// Bandwidth-bound; weights staged in LDS.
// ---------------------------------------------------------------------------
__global__ __launch_bounds__(256) void head_kernel(
    const bf16_t* __restrict__ hs, const float* __restrict__ Wa,
    const float* __restrict__ ba, const float* __restrict__ Wc,
    const float* __restrict__ bc, float* __restrict__ out)
{
  __shared__ float sWa[128 * 6];
  __shared__ float sWc[128];
  __shared__ float sb[8];
  const int tid = threadIdx.x;
  for (int i = tid; i < 768; i += 256) sWa[i] = Wa[i];
  if (tid < 128) sWc[tid] = Wc[tid];
  if (tid < 6)   sb[tid]  = ba[tid];
  if (tid == 6)  sb[6]    = bc[0];
  __syncthreads();

  const size_t row = (size_t)blockIdx.x * 256 + tid;
  if (row >= (size_t)N_ROWS) return;

  float a0 = sb[0], a1 = sb[1], a2 = sb[2], a3 = sb[3], a4 = sb[4], a5 = sb[5];
  float av = sb[6];
  const bf16_t* hp = hs + row * 128;
  for (int kb = 0; kb < 16; ++kb) {
    v8bf hv = *(const v8bf*)(hp + kb * 8);
#pragma unroll
    for (int e = 0; e < 8; ++e) {
      float hk = (float)hv[e];
      const float* w = &sWa[(kb * 8 + e) * 6];
      a0 += hk * w[0]; a1 += hk * w[1]; a2 += hk * w[2];
      a3 += hk * w[3]; a4 += hk * w[4]; a5 += hk * w[5];
      av += hk * sWc[kb * 8 + e];
    }
  }
  float* o = out + row * 7;
  o[0] = a0; o[1] = a1; o[2] = a2; o[3] = a3; o[4] = a4; o[5] = a5; o[6] = av;
}

// ---------------------------------------------------------------------------
// Launch
// ---------------------------------------------------------------------------
extern "C" void kernel_launch(void* const* d_in, const int* in_sizes, int n_in,
                              void* d_out, int out_size, void* d_ws, size_t ws_size,
                              hipStream_t stream)
{
  const float* img = (const float*)d_in[0];
  const float* loc = (const float*)d_in[1];
  const float* eng = (const float*)d_in[2];
  const int*   done= (const int*)  d_in[3];
  const float* h0  = (const float*)d_in[4];
  const float* c0  = (const float*)d_in[5];
  const float* W1  = (const float*)d_in[6];   const float* b1 = (const float*)d_in[7];
  const float* W2  = (const float*)d_in[8];   const float* b2 = (const float*)d_in[9];
  const float* W3  = (const float*)d_in[10];  const float* b3 = (const float*)d_in[11];
  const float* W4  = (const float*)d_in[12];  const float* b4 = (const float*)d_in[13];
  const float* Wl  = (const float*)d_in[14];  const float* bl = (const float*)d_in[15];
  const float* We  = (const float*)d_in[16];  const float* be = (const float*)d_in[17];
  const float* Wih = (const float*)d_in[18];  const float* Whh= (const float*)d_in[19];
  const float* bih = (const float*)d_in[20];  const float* bhh= (const float*)d_in[21];
  const float* Wa  = (const float*)d_in[22];  const float* ba = (const float*)d_in[23];
  const float* Wc  = (const float*)d_in[24];  const float* bc = (const float*)d_in[25];

  // Workspace layout (bytes, 256-aligned segments)
  char* ws = (char*)d_ws;
  bf16_t* W1t   = (bf16_t*)(ws + 0);                         //  16384
  bf16_t* W2t   = (bf16_t*)(ws + 16384);                     // 131072
  bf16_t* W3t   = (bf16_t*)(ws + 147456);                    //  65536
  bf16_t* W4t   = (bf16_t*)(ws + 212992);                    //   8192
  bf16_t* Wihb  = (bf16_t*)(ws + 221184);                    //  98304
  bf16_t* Whhb  = (bf16_t*)(ws + 319488);                    // 131072
  float*  biasg = (float*) (ws + 450560);                    //   2048
  bf16_t* feats = (bf16_t*)(ws + 452608);                    // N*96*2  = 100663296
  float*  hbuf  = (float*) (ws + 101115904);                 // B*128*4 = 2097152
  float*  cbuf  = (float*) (ws + 103213056);                 // B*128*4 = 2097152
  bf16_t* hsbuf = (bf16_t*)(ws + 105310208);                 // N*128*2 = 134217728

  // 1) Weight conversion / transpose (225792 elements total)
  prep_kernel<<<882, 256, 0, stream>>>(W1, W2, W3, W4, Wih, Whh, bih, bhh,
                                       W1t, W2t, W3t, W4t, Wihb, Whhb, biasg);

  // 2) Fused encoder: 8192 tiles of 64 rows
  mlp_kernel<<<N_ROWS / 64, 256, 0, stream>>>(
      img, loc, eng, b1, b2, b3, b4, Wl, bl, We, be,
      W1t, W2t, W3t, W4t, feats);

  // 3) Initial LSTM state
  hipMemcpyAsync(hbuf, h0, (size_t)B_ENV * H_DIM * sizeof(float),
                 hipMemcpyDeviceToDevice, stream);
  hipMemcpyAsync(cbuf, c0, (size_t)B_ENV * H_DIM * sizeof(float),
                 hipMemcpyDeviceToDevice, stream);

  // 4) Sequential LSTM (stream order provides inter-step dependency)
  for (int t = 0; t < T_LEN; ++t) {
    lstm_step_kernel<<<B_ENV / 64, 256, 0, stream>>>(
        t, feats, Wihb, Whhb, biasg, done, hbuf, cbuf, hsbuf);
  }

  // 5) Heads
  head_kernel<<<N_ROWS / 256, 256, 0, stream>>>(hsbuf, Wa, ba, Wc, bc,
                                                (float*)d_out);
}
